// DecoderBlockV1_23957327577635
// MI455X (gfx1250) — compile-verified, matrix operand order checked by reading, not run on hardware
//
#include <hip/hip_runtime.h>

typedef __bf16 bf16;
typedef __attribute__((ext_vector_type(16))) __bf16 v16bf;
typedef __attribute__((ext_vector_type(8)))  float  v8f;
typedef __attribute__((__vector_size__(4 * sizeof(int)))) int v4i_t;

#define HID   1024
#define G4    4096
#define BB    32
#define TSEQ  512
#define OUTD  80
#define NBLK  128
#define NTHR  256
#define NWAVE (NBLK * NTHR / 32)   // 1024 wave32s

// ---------------------------------------------------------------------------
// WMMA fragment helpers (CDNA5 16-bit layouts, CDNA5 ISA §7.12.2)
// A 16x32 bf16: lanes 0-15 = rows, K k0+0..7 & k0+16..23; lanes 16-31 same
// rows, K k0+8..15 & k0+24..31.  B (=W^T) loaded with the identical pattern
// from row-major W[n,k].
// ---------------------------------------------------------------------------
__device__ inline v16bf load_frag(const bf16* __restrict__ rowp, int k0) {
  const int half8 = (threadIdx.x & 16) >> 1;   // 0 for lanes 0-15, 8 for 16-31
  const bf16* p = rowp + k0 + half8;
  v16bf f;
  float4* d = reinterpret_cast<float4*>(&f);
  d[0] = *reinterpret_cast<const float4*>(p);        // 8 bf16
  d[1] = *reinterpret_cast<const float4*>(p + 16);   // 8 bf16
  return f;
}

__device__ inline v8f wmma_bf16(v16bf a, v16bf b, v8f c) {
  return __builtin_amdgcn_wmma_f32_16x16x32_bf16(false, a, false, b,
                                                 (short)0, c, false, false);
}

__device__ inline v8f zero8() { v8f c; for (int i = 0; i < 8; ++i) c[i] = 0.f; return c; }
__device__ inline float sigmoidf_(float x) { return 1.0f / (1.0f + __expf(-x)); }

// ---------------------------------------------------------------------------
// Async Global -> LDS copy (CDNA5 GLOBAL_LOAD_ASYNC_TO_LDS_B128, ASYNCcnt),
// with a plain ds_store fallback if the builtin is unavailable.
// ---------------------------------------------------------------------------
#if __has_builtin(__builtin_amdgcn_global_load_async_to_lds_b128)
#define HAS_ASYNC_LDS 1
#else
#define HAS_ASYNC_LDS 0
#endif

__device__ inline void cp16_to_lds(const void* g, void* l) {
#if HAS_ASYNC_LDS
  __builtin_amdgcn_global_load_async_to_lds_b128(
      (__attribute__((address_space(1))) v4i_t*)g,
      (__attribute__((address_space(3))) v4i_t*)l, 0, 0);
#else
  *(float4*)l = *(const float4*)g;
#endif
}

__device__ inline void cp_to_lds_wait() {
#if HAS_ASYNC_LDS
#if __has_builtin(__builtin_amdgcn_s_wait_asynccnt)
  __builtin_amdgcn_s_wait_asynccnt(0);
#else
  asm volatile("s_wait_asynccnt 0" ::: "memory");
#endif
#endif
}

// ---------------------------------------------------------------------------
// Grid-wide sense-reversal barrier in L2 (device-scope atomics + s_sleep spin)
// ---------------------------------------------------------------------------
__device__ inline void grid_barrier(unsigned* bar) {
  __threadfence();
  __syncthreads();
  if (threadIdx.x == 0) {
    unsigned gen  = __hip_atomic_load(bar + 1, __ATOMIC_RELAXED, __HIP_MEMORY_SCOPE_AGENT);
    unsigned prev = __hip_atomic_fetch_add(bar, 1u, __ATOMIC_ACQ_REL, __HIP_MEMORY_SCOPE_AGENT);
    if (prev == (unsigned)(NBLK - 1)) {
      __hip_atomic_store(bar, 0u, __ATOMIC_RELAXED, __HIP_MEMORY_SCOPE_AGENT);
      __hip_atomic_fetch_add(bar + 1, 1u, __ATOMIC_RELEASE, __HIP_MEMORY_SCOPE_AGENT);
    } else {
      while (__hip_atomic_load(bar + 1, __ATOMIC_ACQUIRE, __HIP_MEMORY_SCOPE_AGENT) == gen)
        __builtin_amdgcn_s_sleep(2);
    }
  }
  __syncthreads();
}

// ---------------------------------------------------------------------------
// Prelude kernels
// ---------------------------------------------------------------------------
__global__ __launch_bounds__(NTHR) void f32_to_bf16_kernel(const float* __restrict__ s,
                                                           bf16* __restrict__ d, int n) {
  int i = blockIdx.x * NTHR + threadIdx.x;
  if (i < n) d[i] = (bf16)s[i];
}

__global__ __launch_bounds__(NTHR) void init_kernel(const float* __restrict__ init_in,
                                                    const float* __restrict__ h0,
                                                    const float* __restrict__ c0,
                                                    float* dec_in, bf16* h_bf,
                                                    float* c_f, unsigned* bar) {
  int i = blockIdx.x * NTHR + threadIdx.x;
  if (i < 2) bar[i] = 0u;
  if (i < BB * OUTD) dec_in[i] = init_in[i];
  if (i < BB * HID) { h_bf[i] = (bf16)h0[i]; c_f[i] = c0[i]; }
}

// K/V projection: one wave per 16x16 tile, K and V share the A fragment.
__global__ __launch_bounds__(NTHR) void kv_gemm_kernel(const bf16* __restrict__ A,
                                                       const bf16* __restrict__ Wk,
                                                       const bf16* __restrict__ Wv,
                                                       const float* __restrict__ bk,
                                                       const float* __restrict__ bv,
                                                       bf16* __restrict__ K,
                                                       bf16* __restrict__ V) {
  const int wave = (blockIdx.x * NTHR + threadIdx.x) >> 5;   // 0..65535
  const int lane = threadIdx.x & 31;
  const int m0 = (wave >> 6) << 4;       // 1024 M tiles (B*T rows)
  const int n0 = (wave & 63) << 4;       // 64 N tiles
  const bf16* arow = A  + (size_t)(m0 + (lane & 15)) * HID;
  const bf16* krow = Wk + (size_t)(n0 + (lane & 15)) * HID;
  const bf16* vrow = Wv + (size_t)(n0 + (lane & 15)) * HID;
  v8f cK = zero8(), cV = zero8();
  for (int k0 = 0; k0 < HID; k0 += 32) {
    v16bf a = load_frag(arow, k0);
    cK = wmma_bf16(a, load_frag(krow, k0), cK);
    cV = wmma_bf16(a, load_frag(vrow, k0), cV);
  }
  const int n = n0 + (lane & 15);
  const int mh = (threadIdx.x & 16) >> 1;
  for (int r = 0; r < 8; ++r) {
    size_t row = (size_t)(m0 + mh + r);
    K[row * HID + n] = (bf16)(cK[r] + bk[n]);
    V[row * HID + n] = (bf16)(cV[r] + bv[n]);
  }
}

// ---------------------------------------------------------------------------
// Persistent decoder
// ---------------------------------------------------------------------------
struct DecParams {
  const float *W1, *b1, *bih, *bhh, *bq, *blin, *W2, *b2;
  const unsigned char* mask;
  const bf16 *Kbf, *Vbf, *Wihbf, *Whhbf, *Wqbf, *Wlbf;
  bf16 *x_bf, *h_bf, *ctx_bf;
  float *c_f, *gatesA, *gatesB, *q_f, *scores, *inv_sum, *blk, *dec_in, *out;
  unsigned* bar;
};

__global__ __launch_bounds__(NTHR) void decode_kernel(DecParams P) {
  const int tid  = threadIdx.x;
  const int gtid = blockIdx.x * NTHR + tid;          // 0..32767
  const int wave = gtid >> 5;                        // 0..1023
  const int lane = tid & 31;
  const float scale = 0.03125f;                      // 1/sqrt(1024)

  // Per-block staged A panel for the gate GEMM: x (blocks 0-63) or h (64-127).
  __shared__ bf16 sA[BB * HID];                      // 64 KB of the 320 KB LDS

  for (int t = 0; t < TSEQ; ++t) {
    // ---- P0: x = relu(dec_in @ W1^T + b1), one output per thread ----------
    {
      const int b = gtid >> 10, j = gtid & (HID - 1);
      const float* w  = P.W1 + (size_t)j * OUTD;
      const float* di = P.dec_in + b * OUTD;
      float acc = P.b1[j];
      for (int k = 0; k < OUTD; ++k) acc = fmaf(di[k], w[k], acc);
      P.x_bf[gtid] = (bf16)fmaxf(acc, 0.0f);
    }
    grid_barrier(P.bar);

    // ---- P1: gates.  Tile (m0,n0) split across two waves: the x@Wih^T half
    //      (waves 0-511 -> gatesA) and the h@Whh^T half (512-1023 -> gatesB).
    //      Each block async-stages its 64KB A panel into LDS first.
    {
      const int half = wave >> 9;                     // 0 = x-part, 1 = h-part
      const bf16* src = half ? P.h_bf : P.x_bf;
      for (int o = tid * 16; o < BB * HID * 2; o += NTHR * 16)
        cp16_to_lds((const char*)src + o, (char*)sA + o);
      cp_to_lds_wait();
      __syncthreads();

      const int tw = wave & 511;
      const int m0 = (tw >> 8) << 4;                  // 2 M tiles
      const int n0 = (tw & 255) << 4;                 // 256 N tiles
      const bf16* arow = sA + (size_t)(m0 + (lane & 15)) * HID;      // LDS
      const bf16* brow = (half ? P.Whhbf : P.Wihbf) +
                         (size_t)(n0 + (lane & 15)) * HID;
      v8f c = zero8();
      for (int k0 = 0; k0 < HID; k0 += 32) {
        __builtin_prefetch(brow + k0 + 64, 0, 0);
        c = wmma_bf16(load_frag(arow, k0), load_frag(brow, k0), c);
      }
      float* dst = half ? P.gatesB : P.gatesA;
      const int n = n0 + (lane & 15);
      const int mh = (tid & 16) >> 1;
      for (int r = 0; r < 8; ++r) dst[(size_t)(m0 + mh + r) * G4 + n] = c[r];
      __syncthreads();                                // protect sA for next step
    }
    grid_barrier(P.bar);

    // ---- P2: LSTM pointwise, one (b,j) per thread -------------------------
    {
      const int b = gtid >> 10, j = gtid & (HID - 1);
      const float* gA = P.gatesA + (size_t)b * G4;
      const float* gB = P.gatesB + (size_t)b * G4;
      float gi = gA[j]           + gB[j]           + P.bih[j]           + P.bhh[j];
      float gf = gA[HID + j]     + gB[HID + j]     + P.bih[HID + j]     + P.bhh[HID + j];
      float gg = gA[2*HID + j]   + gB[2*HID + j]   + P.bih[2*HID + j]   + P.bhh[2*HID + j];
      float go = gA[3*HID + j]   + gB[3*HID + j]   + P.bih[3*HID + j]   + P.bhh[3*HID + j];
      float cn = sigmoidf_(gf) * P.c_f[gtid] + sigmoidf_(gi) * tanhf(gg);
      float hn = sigmoidf_(go) * tanhf(cn);
      P.c_f[gtid] = cn;
      P.h_bf[gtid] = (bf16)hn;
    }
    grid_barrier(P.bar);

    // ---- P3: q = h_new @ Wq^T + bq  (128 tiles) ---------------------------
    if (wave < 128) {
      const int m0 = (wave >> 6) << 4, n0 = (wave & 63) << 4;
      const bf16* hrow = P.h_bf + (size_t)(m0 + (lane & 15)) * HID;
      const bf16* qrow = P.Wqbf + (size_t)(n0 + (lane & 15)) * HID;
      v8f c = zero8();
      for (int k0 = 0; k0 < HID; k0 += 32)
        c = wmma_bf16(load_frag(hrow, k0), load_frag(qrow, k0), c);
      const int n = n0 + (lane & 15);
      const int mh = (tid & 16) >> 1;
      for (int r = 0; r < 8; ++r)
        P.q_f[(size_t)(m0 + mh + r) * HID + n] = c[r] + P.bq[n];
    }
    grid_barrier(P.bar);

    // ---- P4: scores[b,t'] = scale * q[b].K[b,t']  (16 rows per wave) ------
    for (int r = 0; r < 16; ++r) {
      const int row = wave * 16 + r;                 // 0..16383
      const int b = row >> 9, tp = row & (TSEQ - 1);
      const bf16*  kp = P.Kbf + ((size_t)b * TSEQ + tp) * HID;
      const float* qp = P.q_f + (size_t)b * HID;
      float acc = 0.f;
      for (int k = lane; k < HID; k += 32) acc = fmaf((float)kp[k], qp[k], acc);
      for (int off = 16; off > 0; off >>= 1) acc += __shfl_xor(acc, off, 32);
      if (lane == 0)
        P.scores[row] = P.mask[b * TSEQ + tp] ? acc * scale : -1e9f;
    }
    grid_barrier(P.bar);

    // ---- P5: softmax over T per batch (one wave per b) --------------------
    if (wave < BB) {
      float* sc = P.scores + wave * TSEQ;
      float m = -3.4e38f;
      for (int i = lane; i < TSEQ; i += 32) m = fmaxf(m, sc[i]);
      for (int off = 16; off > 0; off >>= 1) m = fmaxf(m, __shfl_xor(m, off, 32));
      float s = 0.f;
      for (int i = lane; i < TSEQ; i += 32) { float e = __expf(sc[i] - m); sc[i] = e; s += e; }
      for (int off = 16; off > 0; off >>= 1) s += __shfl_xor(s, off, 32);
      if (lane == 0) P.inv_sum[wave] = 1.0f / s;
    }
    grid_barrier(P.bar);

    // ---- P6: ctx[b,j] = inv_sum[b] * sum_t attn*V  (32 j per wave) --------
    {
      const int b = wave >> 5;
      const int j = ((wave & 31) << 5) + lane;
      const bf16*  vb = P.Vbf + (size_t)b * TSEQ * HID + j;
      const float* at = P.scores + b * TSEQ;
      float acc = 0.f;
      for (int tp = 0; tp < TSEQ; ++tp) acc = fmaf(at[tp], (float)vb[(size_t)tp * HID], acc);
      P.ctx_bf[b * HID + j] = (bf16)(acc * P.inv_sum[b]);
    }
    grid_barrier(P.bar);

    // ---- P7: blk = ctx @ Wl^T + bl  (128 tiles) ---------------------------
    if (wave < 128) {
      const int m0 = (wave >> 6) << 4, n0 = (wave & 63) << 4;
      const bf16* crow = P.ctx_bf + (size_t)(m0 + (lane & 15)) * HID;
      const bf16* lrow = P.Wlbf  + (size_t)(n0 + (lane & 15)) * HID;
      v8f c = zero8();
      for (int k0 = 0; k0 < HID; k0 += 32)
        c = wmma_bf16(load_frag(crow, k0), load_frag(lrow, k0), c);
      const int n = n0 + (lane & 15);
      const int mh = (tid & 16) >> 1;
      for (int r = 0; r < 8; ++r)
        P.blk[(size_t)(m0 + mh + r) * HID + n] = c[r] + P.blin[n];
    }
    grid_barrier(P.bar);

    // ---- P8: out = blk @ W2^T + b2  (wave per output row, K split) --------
    for (int row = wave; row < BB * OUTD; row += NWAVE) {
      const int b = row / OUTD, o = row - b * OUTD;
      const float* w2 = P.W2 + (size_t)o * HID;
      const float* bf = P.blk + (size_t)b * HID;
      float acc = 0.f;
      for (int k = lane; k < HID; k += 32) acc = fmaf(bf[k], w2[k], acc);
      for (int off = 16; off > 0; off >>= 1) acc += __shfl_xor(acc, off, 32);
      if (lane == 0) {
        float v = acc + P.b2[o];
        P.out[((size_t)b * TSEQ + t) * OUTD + o] = v;
        P.dec_in[b * OUTD + o] = v;
      }
    }
    grid_barrier(P.bar);
  }
}

// ---------------------------------------------------------------------------
// Host side
// ---------------------------------------------------------------------------
extern "C" void kernel_launch(void* const* d_in, const int* in_sizes, int n_in,
                              void* d_out, int out_size, void* d_ws, size_t ws_size,
                              hipStream_t stream) {
  const float* hid_rs = (const float*)d_in[0];
  const unsigned char* mask = (const unsigned char*)d_in[1];
  const float* init_in = (const float*)d_in[2];
  const float* h0 = (const float*)d_in[3];
  const float* c0 = (const float*)d_in[4];
  const float* W1 = (const float*)d_in[5];
  const float* b1 = (const float*)d_in[6];
  const float* Wih = (const float*)d_in[7];
  const float* Whh = (const float*)d_in[8];
  const float* bih = (const float*)d_in[9];
  const float* bhh = (const float*)d_in[10];
  const float* Wq = (const float*)d_in[11];
  const float* bq = (const float*)d_in[12];
  const float* Wk = (const float*)d_in[13];
  const float* bk = (const float*)d_in[14];
  const float* Wv = (const float*)d_in[15];
  const float* bv = (const float*)d_in[16];
  const float* Wl = (const float*)d_in[17];
  const float* bl = (const float*)d_in[18];
  const float* W2 = (const float*)d_in[19];
  const float* b2 = (const float*)d_in[20];

  char* ws = (char*)d_ws;
  size_t off = 0;
  auto alloc = [&](size_t bytes) -> char* {
    char* p = ws + off;
    off += (bytes + 255) & ~(size_t)255;
    return p;
  };
  bf16* hid_bf = (bf16*)alloc((size_t)BB * TSEQ * HID * 2);
  bf16* Kbf    = (bf16*)alloc((size_t)BB * TSEQ * HID * 2);
  bf16* Vbf    = (bf16*)alloc((size_t)BB * TSEQ * HID * 2);
  bf16* Wihbf  = (bf16*)alloc((size_t)G4 * HID * 2);
  bf16* Whhbf  = (bf16*)alloc((size_t)G4 * HID * 2);
  bf16* Wqbf   = (bf16*)alloc((size_t)HID * HID * 2);
  bf16* Wkbf   = (bf16*)alloc((size_t)HID * HID * 2);
  bf16* Wvbf   = (bf16*)alloc((size_t)HID * HID * 2);
  bf16* Wlbf   = (bf16*)alloc((size_t)HID * HID * 2);
  bf16* x_bf   = (bf16*)alloc((size_t)BB * HID * 2);
  bf16* h_bf   = (bf16*)alloc((size_t)BB * HID * 2);
  bf16* ctx_bf = (bf16*)alloc((size_t)BB * HID * 2);
  float* c_f    = (float*)alloc((size_t)BB * HID * 4);
  float* gatesA = (float*)alloc((size_t)BB * G4 * 4);
  float* gatesB = (float*)alloc((size_t)BB * G4 * 4);
  float* q_f    = (float*)alloc((size_t)BB * HID * 4);
  float* scores = (float*)alloc((size_t)BB * TSEQ * 4);
  float* invsum = (float*)alloc((size_t)BB * 4);
  float* blk    = (float*)alloc((size_t)BB * HID * 4);
  float* dec_in = (float*)alloc((size_t)BB * OUTD * 4);
  unsigned* bar = (unsigned*)alloc(256);

  auto cvt = [&](const float* s, bf16* d, int n) {
    f32_to_bf16_kernel<<<(n + NTHR - 1) / NTHR, NTHR, 0, stream>>>(s, d, n);
  };
  cvt(hid_rs, hid_bf, BB * TSEQ * HID);
  cvt(Wih, Wihbf, G4 * HID);
  cvt(Whh, Whhbf, G4 * HID);
  cvt(Wq, Wqbf, HID * HID);
  cvt(Wk, Wkbf, HID * HID);
  cvt(Wv, Wvbf, HID * HID);
  cvt(Wl, Wlbf, HID * HID);

  init_kernel<<<(BB * HID + NTHR - 1) / NTHR, NTHR, 0, stream>>>(
      init_in, h0, c0, dec_in, h_bf, c_f, bar);

  // 65536 tiles (1024 M x 64 N), one wave each, 8 waves/block -> 8192 blocks
  kv_gemm_kernel<<<8192, NTHR, 0, stream>>>(hid_bf, Wkbf, Wvbf, bk, bv, Kbf, Vbf);

  DecParams P;
  P.W1 = W1; P.b1 = b1; P.bih = bih; P.bhh = bhh; P.bq = bq; P.blin = bl;
  P.W2 = W2; P.b2 = b2; P.mask = mask;
  P.Kbf = Kbf; P.Vbf = Vbf; P.Wihbf = Wihbf; P.Whhbf = Whhbf;
  P.Wqbf = Wqbf; P.Wlbf = Wlbf;
  P.x_bf = x_bf; P.h_bf = h_bf; P.ctx_bf = ctx_bf;
  P.c_f = c_f; P.gatesA = gatesA; P.gatesB = gatesB; P.q_f = q_f;
  P.scores = scores; P.inv_sum = invsum; P.blk = blk; P.dec_in = dec_in;
  P.out = (float*)d_out; P.bar = bar;

  decode_kernel<<<NBLK, NTHR, 0, stream>>>(P);
}
